// TransformerEncoderLayer_25907242730072
// MI455X (gfx1250) — matrix operand order repaired and checked by reading, not compiled, not run on hardware
//
#include <hip/hip_runtime.h>
#include <hip/hip_bf16.h>
#include <math.h>

// ---------------------------------------------------------------------------
// CDNA5 (gfx1250) TransformerEncoderLayer, f16 WMMA with f32 accumulation.
// WMMA-only intermediates (qkv, attention out, mlp hidden) stored f16 in HBM.
// Residual/LayerNorm spine kept f32. LDS tiles f16, double-buffered GEMM.
// ---------------------------------------------------------------------------

typedef __attribute__((ext_vector_type(16))) _Float16 v16h;
typedef __attribute__((ext_vector_type(8)))  _Float16 v8h;
typedef __attribute__((ext_vector_type(8)))  float    v8f;

#define NTOK   36864      // B*R*C = 2*576*32
#define N_B    2
#define N_R    576
#define N_C    32

__device__ __forceinline__ v8f vzero8() {
  v8f z;
#pragma unroll
  for (int i = 0; i < 8; ++i) z[i] = 0.0f;
  return z;
}

__device__ __forceinline__ v8f wmma_f16(v16h a, v16h b, v8f c) {
  // D = A(16x32 f16) * B(32x16 f16) + C(16x16 f32)
  return __builtin_amdgcn_wmma_f32_16x16x32_f16(false, a, false, b,
                                                (short)0, c, false, false);
}

// A fragment: 16x32 f16 tile in LDS, row stride ld (halves).
// ISA 7.12.2: lanes 0-15 row M=lane, K={0..7,16..23}; lanes 16-31 K={8..15,24..31}
__device__ __forceinline__ v16h load_a_frag16(const _Float16* p, int ld, int lane) {
  const int r  = lane & 15;
  const int hi = (lane >> 4) & 1;
  const _Float16* row = p + r * ld;
  v8h lo  = *(const v8h*)(row + hi * 8);
  v8h hi8 = *(const v8h*)(row + 16 + hi * 8);
  return __builtin_shufflevector(lo, hi8, 0, 1, 2, 3, 4, 5, 6, 7,
                                 8, 9, 10, 11, 12, 13, 14, 15);
}

// B fragment: 32x16 f16 tile in LDS stored [n][k], row stride ld (halves).
// lanes 0-15: col N=lane, K={0..7,16..23} ; lanes 16-31: K={8..15,24..31}
__device__ __forceinline__ v16h load_b_frag16(const _Float16* p, int ld, int lane) {
  const int n  = lane & 15;
  const int hi = (lane >> 4) & 1;
  const _Float16* row = p + n * ld;
  v8h lo  = *(const v8h*)(row + hi * 8);
  v8h hi8 = *(const v8h*)(row + 16 + hi * 8);
  return __builtin_shufflevector(lo, hi8, 0, 1, 2, 3, 4, 5, 6, 7,
                                 8, 9, 10, 11, 12, 13, 14, 15);
}

// Staging registers: 16 contiguous elements of one tile row per thread.
template <typename T> struct TileReg;
template <> struct TileReg<float> {
  float4 r[4];
  __device__ __forceinline__ void load(const float* p) {
#pragma unroll
    for (int q = 0; q < 4; ++q) r[q] = ((const float4*)p)[q];
  }
  __device__ __forceinline__ void store(_Float16* d) const {
    const float* f = (const float*)r;
#pragma unroll
    for (int j = 0; j < 16; ++j) d[j] = (_Float16)f[j];   // one cvt per element
  }
};
template <> struct TileReg<_Float16> {
  v8h r[2];
  __device__ __forceinline__ void load(const _Float16* p) {
    r[0] = ((const v8h*)p)[0];
    r[1] = ((const v8h*)p)[1];
  }
  __device__ __forceinline__ void store(_Float16* d) const {
    ((v8h*)d)[0] = r[0];
    ((v8h*)d)[1] = r[1];
  }
};

// ---------------------------------------------------------------------------
// GEMM: out[M,N] = (OT) (A[M,K] @ W[N,K]^T + bias (+ residual | gelu))
// A is f32 or f16; W f32; out f32 or f16. Block 256 thr = 8 waves (4x2).
// BM=BN=128, BK=32. LDS f16 tiles, row stride 40 halves, double-buffered.
// EPI: 0 = bias, 1 = bias + residual(f32), 2 = bias + exact GELU
// ---------------------------------------------------------------------------
#define LDT 40            // LDS tile row stride in halves
#define TSZ (128 * LDT)   // halves per tile buffer

template <int EPI, typename AT, typename OT>
__global__ __launch_bounds__(256)
void gemm_f16wmma(const AT* __restrict__ A, const float* __restrict__ W,
                  const float* __restrict__ bias, const float* __restrict__ res,
                  OT* __restrict__ out, int M, int N, int K)
{
  __shared__ __align__(16) _Float16 As[2 * TSZ];
  __shared__ __align__(16) _Float16 Bs[2 * TSZ];

  const int tid  = threadIdx.x;
  const int lane = tid & 31;
  const int wave = tid >> 5;
  const int wm = wave >> 1;           // 0..3 (32 rows each)
  const int wn = wave & 1;            // 0..1 (64 cols each)
  const int m0 = blockIdx.y * 128;
  const int n0 = blockIdx.x * 128;

  v8f acc[2][4];
#pragma unroll
  for (int i = 0; i < 2; ++i)
#pragma unroll
    for (int j = 0; j < 4; ++j) acc[i][j] = vzero8();

  const int lrow = tid >> 1;          // 0..127
  const int lcb  = (tid & 1) * 16;    // 0 or 16

  const AT*    aptr = A + (size_t)(m0 + lrow) * K + lcb;
  const float* wptr = W + (size_t)(n0 + lrow) * K + lcb;
  _Float16* adst = As + lrow * LDT + lcb;
  _Float16* wdst = Bs + lrow * LDT + lcb;

  // Preload + stage tile 0
  TileReg<AT>    ta;
  TileReg<float> tw;
  ta.load(aptr);
  tw.load(wptr);
  ta.store(adst);
  tw.store(wdst);

  int cur = 0;
  for (int k0 = 0; k0 < K; k0 += 32) {
    __syncthreads();
    const bool more = (k0 + 32) < K;
    if (more) {
      ta.load(aptr + k0 + 32);
      tw.load(wptr + k0 + 32);
      __builtin_prefetch(aptr + k0 + 64, 0, 0);
      __builtin_prefetch(wptr + k0 + 64, 0, 0);
    }

    const _Float16* at = As + cur * TSZ;
    const _Float16* bt = Bs + cur * TSZ;
    v16h afrag[2], bfrag[4];
#pragma unroll
    for (int ti = 0; ti < 2; ++ti)
      afrag[ti] = load_a_frag16(at + (wm * 32 + ti * 16) * LDT, LDT, lane);
#pragma unroll
    for (int tj = 0; tj < 4; ++tj)
      bfrag[tj] = load_b_frag16(bt + (wn * 64 + tj * 16) * LDT, LDT, lane);
#pragma unroll
    for (int ti = 0; ti < 2; ++ti)
#pragma unroll
      for (int tj = 0; tj < 4; ++tj)
        acc[ti][tj] = wmma_f16(afrag[ti], bfrag[tj], acc[ti][tj]);

    if (more) {
      ta.store(adst + (cur ^ 1) * TSZ);
      tw.store(wdst + (cur ^ 1) * TSZ);
    }
    cur ^= 1;
  }

  // Epilogue. C/D layout: VGPR i -> M=i (lanes 0-15) / M=8+i (lanes 16-31)
  const int rhalf = (lane >> 4) ? 8 : 0;
  const int cl    = lane & 15;
#pragma unroll
  for (int ti = 0; ti < 2; ++ti) {
#pragma unroll
    for (int tj = 0; tj < 4; ++tj) {
      const int col   = n0 + wn * 64 + tj * 16 + cl;
      const float bv  = bias[col];
      const int rbase = m0 + wm * 32 + ti * 16 + rhalf;
#pragma unroll
      for (int i = 0; i < 8; ++i) {
        const size_t idx = (size_t)(rbase + i) * N + col;
        float v = acc[ti][tj][i] + bv;
        if (EPI == 1) v += res[idx];
        if (EPI == 2) v = 0.5f * v * (1.0f + erff(v * 0.70710678118654752f));
        out[idx] = (OT)v;
      }
    }
  }
}

// ---------------------------------------------------------------------------
// Attention 1: seq len 32, head dim 64. One wave per (batch-of-1152, head).
// qkv (f16) rows: 1536 cols = [Q | K | V], head h at h*64. out f16 [NTOK,512].
// ---------------------------------------------------------------------------
__global__ __launch_bounds__(32)
void attn_small(const _Float16* __restrict__ qkv, _Float16* __restrict__ out)
{
  const int pair = blockIdx.x;         // 1152*8
  const int bb = pair >> 3;
  const int h  = pair & 7;
  const int lane = threadIdx.x;

  __shared__ __align__(16) _Float16 Qs[32 * 64];
  __shared__ __align__(16) _Float16 Ks[32 * 64];
  __shared__ __align__(16) _Float16 Vt[64 * 32];   // [d][key]
  __shared__ __align__(16) _Float16 Ps[32 * 32];   // probabilities
  __shared__ float Ss[32 * 32];                     // f32 scores for softmax

  for (int idx = lane; idx < 32 * 8; idx += 32) {
    const int t = idx >> 3, db = (idx & 7) * 8;
    const size_t ro = (size_t)(bb * 32 + t) * 1536 + h * 64;
    *(v8h*)(Qs + t * 64 + db) = *(const v8h*)(qkv + ro + db);
    *(v8h*)(Ks + t * 64 + db) = *(const v8h*)(qkv + ro + 512 + db);
    v8h v = *(const v8h*)(qkv + ro + 1024 + db);
#pragma unroll
    for (int j = 0; j < 8; ++j) Vt[(db + j) * 32 + t] = v[j];
  }
  __syncthreads();

  // S = Q @ K^T * 1/sqrt(64)
  v8f sacc[2][2];
#pragma unroll
  for (int i = 0; i < 2; ++i)
#pragma unroll
    for (int j = 0; j < 2; ++j) sacc[i][j] = vzero8();

#pragma unroll
  for (int ks = 0; ks < 64; ks += 32) {
    v16h a0 = load_a_frag16(Qs + 0  * 64 + ks, 64, lane);
    v16h a1 = load_a_frag16(Qs + 16 * 64 + ks, 64, lane);
    v16h b0 = load_b_frag16(Ks + 0  * 64 + ks, 64, lane);
    v16h b1 = load_b_frag16(Ks + 16 * 64 + ks, 64, lane);
    sacc[0][0] = wmma_f16(a0, b0, sacc[0][0]);
    sacc[0][1] = wmma_f16(a0, b1, sacc[0][1]);
    sacc[1][0] = wmma_f16(a1, b0, sacc[1][0]);
    sacc[1][1] = wmma_f16(a1, b1, sacc[1][1]);
  }
  const int rhalf = (lane >> 4) ? 8 : 0;
  const int cl    = lane & 15;
#pragma unroll
  for (int ti = 0; ti < 2; ++ti)
#pragma unroll
    for (int tj = 0; tj < 2; ++tj)
#pragma unroll
      for (int i = 0; i < 8; ++i)
        Ss[(ti * 16 + rhalf + i) * 32 + tj * 16 + cl] = sacc[ti][tj][i] * 0.125f;
  __syncthreads();

  // Softmax: one lane per row; write normalized probs as f16 into Ps
  {
    float mx = -1e30f;
#pragma unroll
    for (int c = 0; c < 32; ++c) mx = fmaxf(mx, Ss[lane * 32 + c]);
    float sm = 0.0f;
#pragma unroll
    for (int c = 0; c < 32; ++c) {
      float e = __expf(Ss[lane * 32 + c] - mx);
      Ss[lane * 32 + c] = e;
      sm += e;
    }
    const float inv = 1.0f / sm;
#pragma unroll
    for (int c = 0; c < 32; ++c)
      Ps[lane * 32 + c] = (_Float16)(Ss[lane * 32 + c] * inv);
  }
  __syncthreads();

  // O = P(32x32) @ V(32x64)
  v8f oacc[2][4];
#pragma unroll
  for (int i = 0; i < 2; ++i)
#pragma unroll
    for (int j = 0; j < 4; ++j) oacc[i][j] = vzero8();

#pragma unroll
  for (int ti = 0; ti < 2; ++ti) {
    v16h a = load_a_frag16(Ps + ti * 16 * 32, 32, lane);
#pragma unroll
    for (int tj = 0; tj < 4; ++tj) {
      v16h b = load_b_frag16(Vt + tj * 16 * 32, 32, lane);
      oacc[ti][tj] = wmma_f16(a, b, oacc[ti][tj]);
    }
  }
#pragma unroll
  for (int ti = 0; ti < 2; ++ti)
#pragma unroll
    for (int tj = 0; tj < 4; ++tj)
#pragma unroll
      for (int i = 0; i < 8; ++i) {
        const int row = bb * 32 + ti * 16 + rhalf + i;
        const int col = h * 64 + tj * 16 + cl;
        out[(size_t)row * 512 + col] = (_Float16)oacc[ti][tj][i];
      }
}

// ---------------------------------------------------------------------------
// Attention 2: 64 "batches" (B*C) of Lq=576 queries over Lk=512 train keys.
// Block = 256 thr per (batch, head, 16-query tile). qkv f16, out f16.
// ---------------------------------------------------------------------------
__global__ __launch_bounds__(256)
void attn_big(const _Float16* __restrict__ qkv, _Float16* __restrict__ out)
{
  const int blk   = blockIdx.x;         // 512 pairs * 36 q-tiles
  const int p     = blk / 36;
  const int qt    = blk % 36;
  const int batch = p >> 3;             // 0..63
  const int h     = p & 7;

  __shared__ __align__(16) _Float16 Qs[16 * 64];    //  2 KB
  __shared__ __align__(16) _Float16 KV[128 * 64];   // 16 KB (K chunk, then V^T)
  __shared__ __align__(16) _Float16 Ps[16 * 512];   // 16 KB probabilities
  __shared__ float Ss[16 * 512];                     // 32 KB scores
  __shared__ float Os[16 * 64];                      //  4 KB
  __shared__ float red[16 * 16];
  __shared__ float rowmax[16];
  __shared__ float rowsum[16];

  const int tid  = threadIdx.x;
  const int lane = tid & 31;
  const int wave = tid >> 5;
  const size_t base = (size_t)batch * 576;

  for (int idx = tid; idx < 16 * 8; idx += 256) {
    const int r = idx >> 3, db = (idx & 7) * 8;
    *(v8h*)(Qs + r * 64 + db) =
        *(const v8h*)(qkv + (base + qt * 16 + r) * 1536 + h * 64 + db);
  }

  // ---- Pass 1: S = Q @ K^T, keys staged in 4 chunks of 128 ----
  const int rhalf = (lane >> 4) ? 8 : 0;
  const int cl    = lane & 15;
  for (int kc = 0; kc < 4; ++kc) {
    __syncthreads();
    for (int idx = tid; idx < 128 * 8; idx += 256) {
      const int j = idx >> 3, db = (idx & 7) * 8;
      *(v8h*)(KV + j * 64 + db) =
          *(const v8h*)(qkv + (base + kc * 128 + j) * 1536 + 512 + h * 64 + db);
    }
    __syncthreads();
    v8f sacc = vzero8();
#pragma unroll
    for (int ks = 0; ks < 64; ks += 32) {
      v16h a = load_a_frag16(Qs + ks, 64, lane);
      v16h b = load_b_frag16(KV + (wave * 16) * 64 + ks, 64, lane);
      sacc = wmma_f16(a, b, sacc);
    }
    const int col = kc * 128 + wave * 16 + cl;
#pragma unroll
    for (int i = 0; i < 8; ++i)
      Ss[(rhalf + i) * 512 + col] = sacc[i] * 0.125f;
  }
  __syncthreads();

  // ---- Softmax over 16 x 512: unnormalized exp -> f16 Ps; divide at end ----
  {
    const int row = tid >> 4;           // 0..15
    const int seg = tid & 15;           // 32 cols each
    float mx = -1e30f;
#pragma unroll
    for (int c = 0; c < 32; ++c)
      mx = fmaxf(mx, Ss[row * 512 + seg * 32 + c]);
    red[row * 16 + seg] = mx;
    __syncthreads();
    if (tid < 16) {
      float m = -1e30f;
#pragma unroll
      for (int s = 0; s < 16; ++s) m = fmaxf(m, red[tid * 16 + s]);
      rowmax[tid] = m;
    }
    __syncthreads();
    const float rm = rowmax[row];
    float sm = 0.0f;
#pragma unroll
    for (int c = 0; c < 32; ++c) {
      float e = __expf(Ss[row * 512 + seg * 32 + c] - rm);
      Ps[row * 512 + seg * 32 + c] = (_Float16)e;
      sm += e;
    }
    red[row * 16 + seg] = sm;
    __syncthreads();
    if (tid < 16) {
      float s = 0.0f;
#pragma unroll
      for (int k = 0; k < 16; ++k) s += red[tid * 16 + k];
      rowsum[tid] = s;
    }
  }

  // ---- Pass 2: O = P @ V. wave -> (n-tile = wave&3, k-half = wave>>2) ----
  const int ntile = wave & 3;
  const int khalf = wave >> 2;
  v8f oacc = vzero8();
  for (int kc = 0; kc < 4; ++kc) {
    __syncthreads();
    for (int idx = tid; idx < 128 * 8; idx += 256) {
      const int j = idx >> 3, db = (idx & 7) * 8;
      v8h v = *(const v8h*)(qkv + (base + kc * 128 + j) * 1536 + 1024 + h * 64 + db);
#pragma unroll
      for (int jj = 0; jj < 8; ++jj) KV[(db + jj) * 128 + j] = v[jj];
    }
    __syncthreads();
#pragma unroll
    for (int s = 0; s < 2; ++s) {
      const int ks = khalf * 64 + s * 32;           // key offset within chunk
      v16h a = load_a_frag16(Ps + kc * 128 + ks, 512, lane);
      v16h b = load_b_frag16(KV + (ntile * 16) * 128 + ks, 128, lane);
      oacc = wmma_f16(a, b, oacc);
    }
  }
  __syncthreads();
  {
    const int col = ntile * 16 + cl;
    if (khalf == 0) {
#pragma unroll
      for (int i = 0; i < 8; ++i) Os[(rhalf + i) * 64 + col] = oacc[i];
    }
  }
  __syncthreads();
  {
    const int col = ntile * 16 + cl;
    if (khalf == 1) {
#pragma unroll
      for (int i = 0; i < 8; ++i) Os[(rhalf + i) * 64 + col] += oacc[i];
    }
  }
  __syncthreads();

  for (int idx = tid; idx < 16 * 64; idx += 256) {
    const int r = idx >> 6, d = idx & 63;
    out[(base + qt * 16 + r) * 512 + h * 64 + d] =
        (_Float16)(Os[idx] / rowsum[r]);
  }
}

// ---------------------------------------------------------------------------
// LayerNorm over last dim E=512. One block of 256 per row. f32 in/out.
// ---------------------------------------------------------------------------
__global__ __launch_bounds__(256)
void ln_kernel(const float* __restrict__ in, const float* __restrict__ g,
               const float* __restrict__ b, float* __restrict__ out)
{
  __shared__ float s1[256];
  __shared__ float s2[256];
  const int row = blockIdx.x;
  const int tid = threadIdx.x;
  const float* x = in + (size_t)row * 512;
  const float a0 = x[tid];
  const float a1 = x[tid + 256];
  s1[tid] = a0 + a1;
  s2[tid] = a0 * a0 + a1 * a1;
  __syncthreads();
  for (int off = 128; off > 0; off >>= 1) {
    if (tid < off) {
      s1[tid] += s1[tid + off];
      s2[tid] += s2[tid + off];
    }
    __syncthreads();
  }
  const float mean = s1[0] * (1.0f / 512.0f);
  const float var  = s2[0] * (1.0f / 512.0f) - mean * mean;
  const float rstd = rsqrtf(var + 1e-5f);
  float* o = out + (size_t)row * 512;
  o[tid]       = (a0 - mean) * rstd * g[tid]       + b[tid];
  o[tid + 256] = (a1 - mean) * rstd * g[tid + 256] + b[tid + 256];
}

// ---------------------------------------------------------------------------
// Transpose (B,R,C,E) <-> (B,C,R,E). One 128-thread block per E-row. f32.
// ---------------------------------------------------------------------------
__global__ __launch_bounds__(128)
void transpose_kernel(const float* __restrict__ in, float* __restrict__ out,
                      int fwd)
{
  const int s = blockIdx.x;
  int dst;
  if (fwd) {
    const int b = s / (N_R * N_C);
    const int rc = s % (N_R * N_C);
    const int r = rc >> 5, c = rc & 31;
    dst = (b * N_C + c) * N_R + r;
  } else {
    const int b = s / (N_C * N_R);
    const int cr = s % (N_C * N_R);
    const int c = cr / N_R, r = cr % N_R;
    dst = (b * N_R + r) * N_C + c;
  }
  const float4* src = (const float4*)(in + (size_t)s * 512);
  float4* d = (float4*)(out + (size_t)dst * 512);
  d[threadIdx.x] = src[threadIdx.x];
}

// ---------------------------------------------------------------------------
// Driver. Workspace regions (bytes): A = NTOK*1536*4, B = C = NTOK*512*4.
// Regions are re-typed per stage (f16 qkv/attn-out/h; f32 spine).
// ---------------------------------------------------------------------------
extern "C" void kernel_launch(void* const* d_in, const int* in_sizes, int n_in,
                              void* d_out, int out_size, void* d_ws, size_t ws_size,
                              hipStream_t stream)
{
  (void)in_sizes; (void)n_in; (void)out_size; (void)ws_size;

  const float* x        = (const float*)d_in[0];
  const float* fa_w_in  = (const float*)d_in[1];
  const float* fa_b_in  = (const float*)d_in[2];
  const float* fa_w_out = (const float*)d_in[3];
  const float* fa_b_out = (const float*)d_in[4];
  const float* sa_w_in  = (const float*)d_in[5];
  const float* sa_b_in  = (const float*)d_in[6];
  const float* sa_w_out = (const float*)d_in[7];
  const float* sa_b_out = (const float*)d_in[8];
  const float* mlp_w1   = (const float*)d_in[9];
  const float* mlp_b1   = (const float*)d_in[10];
  const float* mlp_w2   = (const float*)d_in[11];
  const float* mlp_b2   = (const float*)d_in[12];
  const float* g1 = (const float*)d_in[13];
  const float* b1 = (const float*)d_in[14];
  const float* g2 = (const float*)d_in[15];
  const float* b2 = (const float*)d_in[16];
  const float* g3 = (const float*)d_in[17];
  const float* b3 = (const float*)d_in[18];

  char* wsb = (char*)d_ws;
  const size_t szA = (size_t)NTOK * 1536 * 4;
  const size_t szB = (size_t)NTOK * 512 * 4;
  _Float16* bufA_h = (_Float16*)wsb;                 // f16 view of region A
  float*    bufA_f = (float*)wsb;                    // f32 view of region A
  _Float16* bufB_h = (_Float16*)(wsb + szA);
  float*    bufB_f = (float*)(wsb + szA);
  float*    bufC_f = (float*)(wsb + szA + szB);

  const dim3 blk256(256);
  const int MB = NTOK / 128;                         // 288

  // 1. QKV1 (f16) = x @ fa_w_in^T + b
  gemm_f16wmma<0, float, _Float16><<<dim3(12, MB), blk256, 0, stream>>>(
      x, fa_w_in, fa_b_in, nullptr, bufA_h, NTOK, 1536, 512);
  // 2. attention over C=32 -> f16
  attn_small<<<dim3(1152 * 8), dim3(32), 0, stream>>>(bufA_h, bufB_h);
  // 3. out-proj + residual(x) -> f32 bufC
  gemm_f16wmma<1, _Float16, float><<<dim3(4, MB), blk256, 0, stream>>>(
      bufB_h, fa_w_out, fa_b_out, x, bufC_f, NTOK, 512, 512);
  // 4. LN1 -> bufB (f32)
  ln_kernel<<<dim3(NTOK), blk256, 0, stream>>>(bufC_f, g1, b1, bufB_f);
  // 5. transpose (B,R,C,E)->(B,C,R,E) -> bufC (xt, f32)
  transpose_kernel<<<dim3(NTOK), dim3(128), 0, stream>>>(bufB_f, bufC_f, 1);
  // 6. QKV2 (f16) = xt @ sa_w_in^T + b
  gemm_f16wmma<0, float, _Float16><<<dim3(12, MB), blk256, 0, stream>>>(
      bufC_f, sa_w_in, sa_b_in, nullptr, bufA_h, NTOK, 1536, 512);
  // 7. attention: 576 queries x 512 train keys -> f16
  attn_big<<<dim3(512 * 36), blk256, 0, stream>>>(bufA_h, bufB_h);
  // 8. out-proj + residual(xt) -> f32 bufA
  gemm_f16wmma<1, _Float16, float><<<dim3(4, MB), blk256, 0, stream>>>(
      bufB_h, sa_w_out, sa_b_out, bufC_f, bufA_f, NTOK, 512, 512);
  // 9. transpose back -> bufB (f32)
  transpose_kernel<<<dim3(NTOK), dim3(128), 0, stream>>>(bufA_f, bufB_f, 0);
  // 10. LN2 -> bufC (f32)
  ln_kernel<<<dim3(NTOK), blk256, 0, stream>>>(bufB_f, g2, b2, bufC_f);
  // 11. h (f16) = gelu(x2 @ w1^T + b1)
  gemm_f16wmma<2, float, _Float16><<<dim3(8, MB), blk256, 0, stream>>>(
      bufC_f, mlp_w1, mlp_b1, nullptr, bufA_h, NTOK, 1024, 512);
  // 12. x3 = x2 + h @ w2^T + b2 -> bufB (f32)
  gemm_f16wmma<1, _Float16, float><<<dim3(4, MB), blk256, 0, stream>>>(
      bufA_h, mlp_w2, mlp_b2, bufC_f, bufB_f, NTOK, 512, 1024);
  // 13. LN3 -> output
  ln_kernel<<<dim3(NTOK), blk256, 0, stream>>>(bufB_f, g3, b3, (float*)d_out);
}
